// SAGE_64287070486569
// MI455X (gfx1250) — matrix operand order repaired and checked by reading, not compiled, not run on hardware
//
#include <hip/hip_runtime.h>

typedef __attribute__((ext_vector_type(16))) _Float16 v16h;
typedef __attribute__((ext_vector_type(8)))  float    v8f;

// ---------------- utility kernels ----------------

__global__ void sage_zero_f32(float* __restrict__ p, long long n) {
  long long i = (long long)blockIdx.x * blockDim.x + threadIdx.x;
  long long stride = (long long)gridDim.x * blockDim.x;
  for (; i < n; i += stride) p[i] = 0.0f;
}

// One thread handles 4 consecutive features of one edge; chunk 0 also bumps degree.
// cshift = log2(din/4), so e = tid >> cshift (no integer division).
__global__ void sage_edge_agg(const float* __restrict__ h,
                              const int* __restrict__ src,
                              const int* __restrict__ dst,
                              float* __restrict__ sum,
                              float* __restrict__ deg,
                              unsigned int total, int din, int cshift) {
  unsigned int tid = blockIdx.x * blockDim.x + threadIdx.x;
  if (tid >= total) return;
  unsigned int e = tid >> cshift;
  unsigned int c = (tid & ((1u << cshift) - 1u)) << 2;
  int s = src[e];
  int d = dst[e];
  const float4 v = *(const float4*)(h + (size_t)s * din + c);
  float* o = sum + (size_t)d * din + c;
  atomicAdd(o + 0, v.x);
  atomicAdd(o + 1, v.y);
  atomicAdd(o + 2, v.z);
  atomicAdd(o + 3, v.w);
  if (c == 0) atomicAdd(deg + d, 1.0f);
}

// Build A_f16[i, 0:din]    = h[i, :]              (self features)
//       A_f16[i, din:2din] = sum[i, :]/max(deg,1) (mean neighbor features)
__global__ void sage_prep_A(const float* __restrict__ h,
                            const float* __restrict__ sum,
                            const float* __restrict__ deg,
                            _Float16* __restrict__ A,
                            long long total, int din, int dshift) {
  long long tid = (long long)blockIdx.x * blockDim.x + threadIdx.x;
  if (tid >= total) return;
  long long i = tid >> dshift;
  int k = (int)(tid & (din - 1));
  float dg = fmaxf(deg[i], 1.0f);
  size_t arow = (size_t)i * ((size_t)din * 2);
  A[arow + k]       = (_Float16)h[(size_t)i * din + k];
  A[arow + din + k] = (_Float16)(sum[(size_t)i * din + k] / dg);
}

// Build W = [Ws; Wn] (K = 2*din rows, Npad cols, zero pad cols >= dout) directly
// in WMMA B-fragment order so the GEMM reads each fragment as one contiguous
// 32-byte per-lane load:
//   Wf[((kt*nNt + nt)*32 + lane)*16 + 2p .. +1] = W[kt*32 + (lane>>4)*16 + 2p (+1)][nt*16 + (lane&15)]
__global__ void sage_prep_Wfrag(const float* __restrict__ Ws,
                                const float* __restrict__ Wn,
                                _Float16* __restrict__ Wf,
                                int din, int dout, int Npad) {
  const int K = 2 * din;
  const int nKt = K >> 5;
  const int nNt = Npad >> 4;
  const int total = nKt * nNt * 32 * 8;  // one thread per half-pair
  int tid = blockIdx.x * blockDim.x + threadIdx.x;
  if (tid >= total) return;
  const int p    = tid & 7;
  const int lane = (tid >> 3) & 31;
  const int frag = tid >> 8;          // kt * nNt + nt
  const int nt   = frag % nNt;
  const int kt   = frag / nNt;
  const int k = (kt << 5) + ((lane >> 4) << 4) + (p << 1);
  const int n = (nt << 4) + (lane & 15);
  float v0 = 0.0f, v1 = 0.0f;
  if (n < dout) {
    v0 = (k < din) ? Ws[(size_t)k * dout + n] : Wn[(size_t)(k - din) * dout + n];
    const int k1 = k + 1;
    v1 = (k1 < din) ? Ws[(size_t)k1 * dout + n] : Wn[(size_t)(k1 - din) * dout + n];
  }
  size_t base = ((size_t)frag * 32 + lane) * 16 + (p << 1);
  Wf[base]     = (_Float16)v0;
  Wf[base + 1] = (_Float16)v1;
}

// ---------------- WMMA GEMM ----------------
// One wave per 16x64 output tile: A fragment loaded once per K-step (2x b128),
// reused by 4 back-to-back v_wmma_f32_16x16x32_f16; B fragments are single
// contiguous 32B per-lane loads from the pre-swizzled Wf.
// Grid covers tiles exactly -> EXEC all-1s at every WMMA (ISA 7.12 restriction).
__global__ void sage_wmma_gemm(const _Float16* __restrict__ A,
                               const _Float16* __restrict__ Wf,
                               const float* __restrict__ bias,
                               float* __restrict__ out,
                               int K, int Npad, int dout, int ntileN4, int relu) {
  const int tile = blockIdx.x;
  const int row0 = (tile / ntileN4) << 4;
  const int ng   = tile % ntileN4;     // group of 64 columns
  const int nNt  = Npad >> 4;
  const int lane = threadIdx.x & 31;
  const int mr = lane & 15;            // A row / B,C column within tile
  const int kh = (lane >> 4) & 1;      // lane half -> K group / C row group

  v8f acc0 = {}, acc1 = {}, acc2 = {}, acc3 = {};
  const int nKt = K >> 5;
  for (int kt = 0; kt < nKt; ++kt) {
    // A fragment: one contiguous 16-half (32B) chunk per lane, k = 16*kh + s
    const _Float16* pa = A + (size_t)(row0 + mr) * K + (kt << 5) + (kh << 4);
    v16h af;
#pragma unroll
    for (int j = 0; j < 8; ++j) { af[j] = pa[j]; af[8 + j] = pa[8 + j]; }

    const _Float16* pb = Wf + ((size_t)(kt * nNt + (ng << 2)) * 32 + lane) * 16;
    v16h b0 = *(const v16h*)(pb);
    v16h b1 = *(const v16h*)(pb + 32 * 16);
    v16h b2 = *(const v16h*)(pb + 2 * 32 * 16);
    v16h b3 = *(const v16h*)(pb + 3 * 32 * 16);
    acc0 = __builtin_amdgcn_wmma_f32_16x16x32_f16(false, af, false, b0, (short)0, acc0, false, false);
    acc1 = __builtin_amdgcn_wmma_f32_16x16x32_f16(false, af, false, b1, (short)0, acc1, false, false);
    acc2 = __builtin_amdgcn_wmma_f32_16x16x32_f16(false, af, false, b2, (short)0, acc2, false, false);
    acc3 = __builtin_amdgcn_wmma_f32_16x16x32_f16(false, af, false, b3, (short)0, acc3, false, false);
  }

  v8f accs[4] = {acc0, acc1, acc2, acc3};
#pragma unroll
  for (int t = 0; t < 4; ++t) {
    const int col = (ng << 6) + (t << 4) + mr;
    if (col < dout) {
      const float bv = bias[col];
#pragma unroll
      for (int r = 0; r < 8; ++r) {
        const int row = row0 + (kh << 3) + r;
        float v = accs[t][r] + bv;
        if (relu) v = fmaxf(v, 0.0f);
        out[(size_t)row * dout + col] = v;
      }
    }
  }
}

// ---------------- host orchestration ----------------

static inline int iceil(long long a, int b) { return (int)((a + b - 1) / b); }

extern "C" void kernel_launch(void* const* d_in, const int* in_sizes, int n_in,
                              void* d_out, int out_size, void* d_ws, size_t ws_size,
                              hipStream_t stream) {
  (void)n_in; (void)out_size; (void)ws_size;

  const float* x        = (const float*)d_in[0];
  const int*   srcp[3]  = {(const int*)d_in[1], (const int*)d_in[3], (const int*)d_in[5]};
  const int*   dstp[3]  = {(const int*)d_in[2], (const int*)d_in[4], (const int*)d_in[6]};
  const float* Wsp[3]   = {(const float*)d_in[7],  (const float*)d_in[10], (const float*)d_in[13]};
  const float* Wnp[3]   = {(const float*)d_in[8],  (const float*)d_in[11], (const float*)d_in[14]};
  const float* bp[3]    = {(const float*)d_in[9],  (const float*)d_in[12], (const float*)d_in[15]};
  const int    E[3]     = {in_sizes[1], in_sizes[3], in_sizes[5]};

  const int nDst[3] = {120000, 20000, 4000};
  const int din[3]  = {128, 256, 256};
  const int dout[3] = {256, 256, 47};
  const int Npad[3] = {256, 256, 64};   // N padded to multiple of 64 (wave tile width)

  // workspace carve (deterministic offsets, 256B aligned)
  char* ws = (char*)d_ws;
  size_t off = 0;
  auto carve = [&](size_t bytes) -> char* {
    char* p = ws + off;
    off += (bytes + 255) & ~(size_t)255;
    return p;
  };
  float*    h1  = (float*)carve((size_t)120000 * 256 * sizeof(float));
  float*    h2  = (float*)carve((size_t)20000  * 256 * sizeof(float));
  float*    sum = (float*)carve((size_t)120000 * 128 * sizeof(float)); // max across layers
  float*    deg = (float*)carve((size_t)120000 * sizeof(float));
  _Float16* Af  = (_Float16*)carve((size_t)120000 * 256 * sizeof(_Float16)); // max across layers
  _Float16* Wf  = (_Float16*)carve((size_t)512 * 256 * sizeof(_Float16));    // max across layers

  const float* hin = x;
  float* outs[3] = {h1, h2, (float*)d_out};

  for (int l = 0; l < 3; ++l) {
    const int nd = nDst[l], di = din[l], dox = dout[l], np = Npad[l];
    const int Ktot = 2 * di;
    const int cshift = (di == 128) ? 5 : 6;  // log2(din/4)
    const int dshift = (di == 128) ? 7 : 8;  // log2(din)

    // 1) zero accumulator + degree
    {
      long long n = (long long)nd * di;
      sage_zero_f32<<<dim3(iceil(n, 256 * 8)), dim3(256), 0, stream>>>(sum, n);
      sage_zero_f32<<<dim3(iceil(nd, 256 * 8)), dim3(256), 0, stream>>>(deg, nd);
    }
    // 2) edge aggregation (L2-resident float atomics)
    {
      unsigned int total = (unsigned int)((long long)E[l] << cshift);
      sage_edge_agg<<<dim3(iceil((long long)total, 256)), dim3(256), 0, stream>>>(
          hin, srcp[l], dstp[l], sum, deg, total, di, cshift);
    }
    // 3) build fused f16 operands (A row-major, W pre-swizzled to fragment order)
    {
      long long totalA = (long long)nd * di;
      sage_prep_A<<<dim3(iceil(totalA, 256)), dim3(256), 0, stream>>>(
          hin, sum, deg, Af, totalA, di, dshift);
      int totalW = (Ktot >> 5) * (np >> 4) * 32 * 8;
      sage_prep_Wfrag<<<dim3(iceil(totalW, 256)), dim3(256), 0, stream>>>(
          Wsp[l], Wnp[l], Wf, di, dox, np);
    }
    // 4) fused dual-GEMM via WMMA: out = [h | mean] @ [Ws; Wn] + b (+ReLU)
    {
      const int ntileN4 = np >> 6;
      const int ntiles = (nd >> 4) * ntileN4;
      sage_wmma_gemm<<<dim3(ntiles), dim3(32), 0, stream>>>(
          Af, Wf, bp[l], outs[l], Ktot, np, dox, ntileN4, (l < 2) ? 1 : 0);
    }
    hin = outs[l];
  }
}
